// DreamWorld_2310692405372
// MI455X (gfx1250) — matrix operand order repaired and checked by reading, not compiled, not run on hardware
//
#include <hip/hip_runtime.h>
#include <hip/hip_bf16.h>

typedef __attribute__((ext_vector_type(16))) _Float16 v16h;
typedef __attribute__((ext_vector_type(8)))  float    v8f;
typedef __attribute__((ext_vector_type(4)))  float    v4f;
typedef __attribute__((ext_vector_type(2)))  float    v2f;
typedef __attribute__((ext_vector_type(4)))  int      v4i;

// -------- workspace layout (in _Float16 elements) --------
// W1   : 16 pos x 2 chan-groups x 3 K-chunks tiles of 512  (conv3x3 effective)
// W2   : 1 tile (w_start2, M=16 out, K=32 in)
// OB1  : 1 tile (w_ob1,  K=16 padded to 32)
// OB2  : 1 tile (w_ob2,  K=20 padded, M=8 padded to 16)
// RG   : 1 tile (w_reward rows 0..7, w_go rows 8..15, K=16 padded)
// FCR  : 4 tiles (fc_reward_w, K=128 as 4 chunks, M=4 padded)
// FCG  : 4 tiles (fc_go_w)
#define OFF_W1   0
#define OFF_W2   49152
#define OFF_OB1  49664
#define OFF_OB2  50176
#define OFF_RG   50688
#define OFF_FCR  51200
#define OFF_FCG  53248
#define WS_TOTAL 55296

union V16H {
    v16h v;
    _Float16 h[16];
    unsigned u[8];
};

static __device__ __forceinline__ v8f wmma_f16(v16h a, v16h b, v8f c) {
    return __builtin_amdgcn_wmma_f32_16x16x32_f16(false, a, false, b, (short)0, c,
                                                  false, false);
}

// lane ^ 16 cross-half swap as pure VALU (v_permlanex16_b32, identity selects)
static __device__ __forceinline__ unsigned xor16_u(unsigned x) {
#if __has_builtin(__builtin_amdgcn_permlanex16)
    return (unsigned)__builtin_amdgcn_permlanex16((int)x, (int)x, 0x76543210,
                                                  (int)0xfedcba98u, false, false);
#else
    return (unsigned)__shfl_xor((int)x, 16, 32);
#endif
}
static __device__ __forceinline__ float xor16_f(float x) {
    union { float f; unsigned u; } c;
    c.f = x;
    c.u = xor16_u(c.u);
    return c.f;
}

// A-operand (16x32 f16) tile: lane l, half h -> m = l%16,
// k = h + (l<16 ? 0 : 16). Tiles stored as [lane][16 halfs].
static __device__ __forceinline__ v16h ldA(const _Float16* base, int lane) {
    return *(const v16h*)(base + lane * 16);
}

// ---------------- prep kernel: build f16 weight tables ----------------
__global__ void prep_weights(const float* __restrict__ w_start1,
                             const float* __restrict__ b_start1,
                             const float* __restrict__ w_start2,
                             const float* __restrict__ w_ob1,
                             const float* __restrict__ w_ob2,
                             const float* __restrict__ w_reward,
                             const float* __restrict__ fc_reward_w,
                             const float* __restrict__ w_go,
                             const float* __restrict__ fc_go_w,
                             _Float16* __restrict__ ws) {
    int idx = blockIdx.x * blockDim.x + threadIdx.x;
    if (idx >= WS_TOTAL) return;
    float val = 0.0f;
    if (idx < OFF_W2) {
        // conv3x3 effective weights (board one-hot dims 0..63, action dims 64..67)
        int t = idx >> 9, r = idx & 511;
        int l = r >> 4, h = r & 15;
        int m = l & 15;
        int k = h + ((l < 16) ? 0 : 16);
        int p = t / 6, grp = (t / 3) & 1, chunk = t % 3;
        int c = grp * 16 + m;
        int kk = chunk * 32 + k;
        int y = p >> 2, x = p & 3;
        if (kk < 64) {
            int pin = kk >> 2, piece = kk & 3;
            int dy = (pin >> 2) - y + 1, dx = (pin & 3) - x + 1;
            if (dy >= 0 && dy < 3 && dx >= 0 && dx < 3)
                val = w_start1[c * 72 + piece * 9 + dy * 3 + dx];
        } else if (kk < 68) {
            int a = kk - 64;
            val = b_start1[c];
            for (int dy = 0; dy < 3; ++dy)
                for (int dx = 0; dx < 3; ++dx) {
                    int yi = y + dy - 1, xi = x + dx - 1;
                    if (yi >= 0 && yi < 4 && xi >= 0 && xi < 4)
                        val += w_start1[c * 72 + (4 + a) * 9 + dy * 3 + dx];
                }
        }
    } else {
        int j = idx - OFF_W2;
        int tile = j >> 9, r = j & 511;
        int l = r >> 4, h = r & 15;
        int m = l & 15;
        int k = h + ((l < 16) ? 0 : 16);
        if (tile == 0) {
            val = w_start2[m * 32 + k];
        } else if (tile == 1) {
            if (k < 16) val = w_ob1[m * 16 + k];
        } else if (tile == 2) {
            if (m < 8 && k < 20) val = w_ob2[m * 20 + k];
        } else if (tile == 3) {
            if (k < 16) val = (m < 8) ? w_reward[m * 16 + k] : w_go[(m - 8) * 16 + k];
        } else if (tile < 8) {
            int kc = tile - 4;
            if (m < 4) val = fc_reward_w[m * 128 + kc * 32 + k];
        } else {
            int kc = tile - 8;
            if (m < 4) val = fc_go_w[m * 128 + kc * 32 + k];
        }
    }
    ws[idx] = (_Float16)val;
}

// ---------------- main fused kernel: ONE 16-row tile per wave ----------------
__global__ __launch_bounds__(256, 1) void dreamworld_main(
    const int* __restrict__ ob, const int* __restrict__ ac,
    const _Float16* __restrict__ ws,
    const float* __restrict__ b_start2, const float* __restrict__ b_ob1,
    const float* __restrict__ b_ob2, const float* __restrict__ b_reward,
    const float* __restrict__ fc_reward_b, const float* __restrict__ b_go,
    const float* __restrict__ fc_go_b, float* __restrict__ out, int B) {
    const int lane = threadIdx.x & 31;
    const bool islo = lane < 16;
    const int tile = (blockIdx.x * blockDim.x + threadIdx.x) >> 5;
    const int ntiles = B >> 4;
    if (tile >= ntiles) return;

    float* __restrict__ out_loc   = out;
    float* __restrict__ out_scale = out + (size_t)B * 64;
    float* __restrict__ out_rl    = out + (size_t)B * 128;
    float* __restrict__ out_rs    = out + (size_t)B * 130;
    float* __restrict__ out_gl    = out + (size_t)B * 132;
    float* __restrict__ out_gs    = out + (size_t)B * 134;

    // per-lane biases (C layout: n = lane%16 batch, m = v + 8*(lane>=16) out chan)
    const int dlt = islo ? 0 : 8;
    float bw2[8], bob1v[8], bob2v[8], brg[8];
#pragma unroll
    for (int v = 0; v < 8; ++v) {
        bw2[v]   = b_start2[v + dlt];
        bob1v[v] = b_ob1[v + dlt];
        bob2v[v] = islo ? b_ob2[v] : 0.0f;
        brg[v]   = islo ? b_reward[v] : b_go[v];
    }

    const int row = (tile << 4) + (lane & 15);  // batch element for this lane

    v4i ob16[4];
#pragma unroll
    for (int q = 0; q < 4; ++q)
        ob16[q] = *(const v4i*)(ob + (size_t)row * 16 + q * 4);
    const int acv = ac[row];

    // ---- one-hot B operands (K chunks of the conv3x3 GEMM), built once ----
    // B layout: lane<16 -> k = chunk*32 + h ; lane>=16 -> k = chunk*32 + 16 + h
    V16H s0, s1, s2;
    {
        v4i qa = islo ? ob16[0] : ob16[1];  // positions 0..3 / 4..7
        v4i qb = islo ? ob16[2] : ob16[3];  // positions 8..11 / 12..15
#pragma unroll
        for (int j = 0; j < 4; ++j) {
            int pa = qa[j], pb = qb[j];
#pragma unroll
            for (int t = 0; t < 4; ++t) {
                s0.h[j * 4 + t] = (pa == t) ? (_Float16)1.0f : (_Float16)0.0f;
                s1.h[j * 4 + t] = (pb == t) ? (_Float16)1.0f : (_Float16)0.0f;
            }
        }
#pragma unroll
        for (int t = 0; t < 16; ++t) s2.h[t] = (_Float16)0.0f;
#pragma unroll
        for (int t = 0; t < 4; ++t)
            s2.h[t] = (islo && acv == t) ? (_Float16)1.0f : (_Float16)0.0f;
    }

    V16H rg8[8];  // reward(lo)/go(hi) activations: chan v, halfs = positions

#pragma unroll
    for (int p = 0; p < 16; ++p) {
        const _Float16* w1p = ws + OFF_W1 + (size_t)(p * 6) * 512;
        // conv3x3: channel groups 0..15 / 16..31, 3 K-chunks each
        v8f alo = {}, ahi = {};
        alo = wmma_f16(ldA(w1p + 0 * 512, lane), s0.v, alo);
        ahi = wmma_f16(ldA(w1p + 3 * 512, lane), s0.v, ahi);
        alo = wmma_f16(ldA(w1p + 1 * 512, lane), s1.v, alo);
        ahi = wmma_f16(ldA(w1p + 4 * 512, lane), s1.v, ahi);
        alo = wmma_f16(ldA(w1p + 2 * 512, lane), s2.v, alo);
        ahi = wmma_f16(ldA(w1p + 5 * 512, lane), s2.v, ahi);

        // relu -> B operand for conv1x1 (K=32 chans), cross-half via permlanex16
        V16H bh;
#pragma unroll
        for (int v = 0; v < 8; ++v) {
            float l0 = fmaxf(alo[v], 0.0f), h0 = fmaxf(ahi[v], 0.0f);
            float lx = xor16_f(l0);
            float hx = xor16_f(h0);
            bh.h[v]     = (_Float16)(islo ? l0 : hx);
            bh.h[v + 8] = (_Float16)(islo ? lx : h0);
        }
        v8f xc = {};
        xc = wmma_f16(ldA(ws + OFF_W2, lane), bh.v, xc);

        // x = relu(.+b); B operand for K=16 layers (upper K half zero)
        float xv[8];
        V16H bx;
#pragma unroll
        for (int v = 0; v < 8; ++v) {
            xv[v] = fmaxf(xc[v] + bw2[v], 0.0f);
            float xx = xor16_f(xv[v]);
            bx.h[v]     = islo ? (_Float16)xv[v] : (_Float16)0.0f;
            bx.h[v + 8] = islo ? (_Float16)xx : (_Float16)0.0f;
        }

        // nol = relu(conv(x, w_ob1)); concat ob one-hot into K dims 16..19
        v8f ncc = {};
        ncc = wmma_f16(ldA(ws + OFF_OB1, lane), bx.v, ncc);
        const int obq = ob16[p >> 2][p & 3];
        V16H bn;
#pragma unroll
        for (int v = 0; v < 8; ++v) {
            float nv = fmaxf(ncc[v] + bob1v[v], 0.0f);
            float nx = xor16_f(nv);
            _Float16 oh = (v < 4 && obq == v) ? (_Float16)1.0f : (_Float16)0.0f;
            bn.h[v]     = islo ? (_Float16)nv : oh;
            bn.h[v + 8] = islo ? (_Float16)nx : (_Float16)0.0f;
        }
        v8f oc = {};
        oc = wmma_f16(ldA(ws + OFF_OB2, lane), bn.v, oc);
        if (islo) {
            v4f f0, f1;
#pragma unroll
            for (int v = 0; v < 4; ++v) {
                f0[v] = oc[v] + bob2v[v];
                f1[v] = oc[v + 4] + bob2v[v + 4];
            }
            *(v4f*)(out_loc + (size_t)row * 64 + p * 4)   = f0;
            *(v4f*)(out_scale + (size_t)row * 64 + p * 4) = f1;
        }

        // fused reward/go 1x1 conv (reward chans in low half, go in high)
        v8f rc = {};
        rc = wmma_f16(ldA(ws + OFF_RG, lane), bx.v, rc);
#pragma unroll
        for (int v = 0; v < 8; ++v)
            rg8[v].h[p] = (_Float16)fmaxf(rc[v] + brg[v], 0.0f);
    }

    // ---- FCs: K=128 = 4 chunks; feature f = c*16 + p ----
    v8f fr = {};
#pragma unroll
    for (int kk = 0; kk < 4; ++kk) {
        V16H bb;
#pragma unroll
        for (int d = 0; d < 8; ++d) {
            unsigned sx = xor16_u(rg8[2 * kk + 1].u[d]);
            bb.u[d] = islo ? rg8[2 * kk].u[d] : sx;
        }
        fr = wmma_f16(ldA(ws + OFF_FCR + kk * 512, lane), bb.v, fr);
    }
    v8f fg = {};
#pragma unroll
    for (int kk = 0; kk < 4; ++kk) {
        V16H bb;
#pragma unroll
        for (int d = 0; d < 8; ++d) {
            unsigned sx = xor16_u(rg8[2 * kk].u[d]);
            bb.u[d] = islo ? sx : rg8[2 * kk + 1].u[d];
        }
        fg = wmma_f16(ldA(ws + OFF_FCG + kk * 512, lane), bb.v, fg);
    }
    if (islo) {
        float frb0 = fc_reward_b[0], frb1 = fc_reward_b[1];
        float frb2 = fc_reward_b[2], frb3 = fc_reward_b[3];
        float fgb0 = fc_go_b[0], fgb1 = fc_go_b[1];
        float fgb2 = fc_go_b[2], fgb3 = fc_go_b[3];
        v2f a, b, c, d;
        a[0] = fr[0] + frb0; a[1] = fr[1] + frb1;
        b[0] = fr[2] + frb2; b[1] = fr[3] + frb3;
        c[0] = fg[0] + fgb0; c[1] = fg[1] + fgb1;
        d[0] = fg[2] + fgb2; d[1] = fg[3] + fgb3;
        *(v2f*)(out_rl + (size_t)row * 2) = a;
        *(v2f*)(out_rs + (size_t)row * 2) = b;
        *(v2f*)(out_gl + (size_t)row * 2) = c;
        *(v2f*)(out_gs + (size_t)row * 2) = d;
    }
}

extern "C" void kernel_launch(void* const* d_in, const int* in_sizes, int n_in,
                              void* d_out, int out_size, void* d_ws, size_t ws_size,
                              hipStream_t stream) {
    const int*   ob          = (const int*)d_in[0];
    const int*   ac          = (const int*)d_in[1];
    const float* w_start1    = (const float*)d_in[2];
    const float* b_start1    = (const float*)d_in[3];
    const float* w_start2    = (const float*)d_in[4];
    const float* b_start2    = (const float*)d_in[5];
    const float* w_ob1       = (const float*)d_in[6];
    const float* b_ob1       = (const float*)d_in[7];
    const float* w_ob2       = (const float*)d_in[8];
    const float* b_ob2       = (const float*)d_in[9];
    const float* w_reward    = (const float*)d_in[10];
    const float* b_reward    = (const float*)d_in[11];
    const float* fc_reward_w = (const float*)d_in[12];
    const float* fc_reward_b = (const float*)d_in[13];
    const float* w_go        = (const float*)d_in[14];
    const float* b_go        = (const float*)d_in[15];
    const float* fc_go_w     = (const float*)d_in[16];
    const float* fc_go_b     = (const float*)d_in[17];
    const int B = in_sizes[1];  // ac has B elements

    _Float16* ws = (_Float16*)d_ws;
    prep_weights<<<(WS_TOTAL + 255) / 256, 256, 0, stream>>>(
        w_start1, b_start1, w_start2, w_ob1, w_ob2, w_reward, fc_reward_w, w_go,
        fc_go_w, ws);

    const int ntiles = B >> 4;                 // 16 batch rows per wave
    const int blocks = (ntiles + 7) / 8;       // 8 waves (256 threads) per block
    dreamworld_main<<<blocks, 256, 0, stream>>>(ob, ac, ws, b_start2, b_ob1, b_ob2,
                                                b_reward, fc_reward_b, b_go, fc_go_b,
                                                (float*)d_out, B);
}
